// AttentionLayer_53884659696274
// MI455X (gfx1250) — matrix operand order repaired and checked by reading, not compiled, not run on hardware
//
#include <hip/hip_runtime.h>

typedef __attribute__((ext_vector_type(16))) _Float16 v16h;
typedef __attribute__((ext_vector_type(8)))  _Float16 v8h;
typedef __attribute__((ext_vector_type(8)))  float    v8f;

union F16x16 { v16h v; v8h h[2]; };

#define B_   4
#define N_   4096
#define D_   128
#define NROW (B_*N_)      // 16384 total rows across batches
#define WT_STRIDE 136     // padded LDS stride (halves) for W^T: breaks 16-way bank conflicts
#define PS 40             // padded LDS stride (halves) for P tiles
#define ROWT 4            // row-tiles per projection block (amortize W staging)

// ---------------------------------------------------------------------------
// Projection: Q = h_a*Wq+bq, K = h_a*Wk+bk, V = h_a*Wv+bv
// grid = (NROW/64, 3); block = 256 (8 waves, one 16x16 column tile each,
// iterated over 4 row-tiles so the LDS W^T staging is amortized 4x)
// y==0 -> Qh [row][d] f16 ; y==1 -> Kh [row][d] f16 ; y==2 -> Vt [b][d][m] f16
// ---------------------------------------------------------------------------
__global__ __launch_bounds__(256)
void proj_qkv_kernel(const float* __restrict__ h_a,
                     const float* __restrict__ Wq, const float* __restrict__ bq,
                     const float* __restrict__ Wk, const float* __restrict__ bk,
                     const float* __restrict__ Wv, const float* __restrict__ bv,
                     _Float16* __restrict__ Qh, _Float16* __restrict__ Kh,
                     _Float16* __restrict__ Vt)
{
    __shared__ _Float16 wlds[128 * WT_STRIDE];
    const int tid  = threadIdx.x;
    const int wid  = tid >> 5;
    const int lane = tid & 31;
    const int col  = lane & 15;
    const int hi   = lane >> 4;
    const int y    = blockIdx.y;

    const float* W    = (y == 0) ? Wq : (y == 1) ? Wk : Wv;
    const float* bias = (y == 0) ? bq : (y == 1) ? bk : bv;

    // Stage W^T (f16) into LDS: wlds[n*WT_STRIDE + k] = W[k][n]
    for (int i = 0; i < 64; ++i) {
        int idx = tid + i * 256;
        int k = idx >> 7, n = idx & 127;
        wlds[n * WT_STRIDE + k] = (_Float16)W[idx];
    }
    __syncthreads();

    const int n0   = wid * 16;             // this wave's output-column tile
    const float bval = bias[n0 + col];

    // B fragments from LDS are loop-invariant across row-tiles: load once.
    F16x16 bfr[4];
    #pragma unroll
    for (int kc = 0; kc < 4; ++kc) {
        const _Float16* bp = wlds + (n0 + col) * WT_STRIDE + kc * 32 + hi * 8;
        bfr[kc].h[0] = *(const v8h*)bp;
        bfr[kc].h[1] = *(const v8h*)(bp + 16);
    }

    for (int rt = 0; rt < ROWT; ++rt) {
        const int rowbase = blockIdx.x * (16 * ROWT) + rt * 16;

        // A fragments (16x32 f16) from fp32 h_a rows (convert on the fly)
        F16x16 afr[4];
        const float* arow = h_a + (size_t)(rowbase + col) * D_;
        #pragma unroll
        for (int kc = 0; kc < 4; ++kc) {
            const float* ap = arow + kc * 32 + hi * 8;
            #pragma unroll
            for (int j = 0; j < 8; ++j) {
                afr[kc].h[0][j] = (_Float16)ap[j];       // k = 8*hi + 0..7
                afr[kc].h[1][j] = (_Float16)ap[16 + j];  // k = 8*hi + 16..23
            }
        }

        v8f acc = {};
        #pragma unroll
        for (int kc = 0; kc < 4; ++kc)
            acc = __builtin_amdgcn_wmma_f32_16x16x32_f16(false, afr[kc].v, false, bfr[kc].v,
                                                         (short)0, acc, false, false);

        #pragma unroll
        for (int r = 0; r < 8; ++r) acc[r] += bval;

        if (y < 2) {
            _Float16* dst = (y == 0) ? Qh : Kh;
            #pragma unroll
            for (int r = 0; r < 8; ++r) {
                int rg = rowbase + r + 8 * hi;
                dst[(size_t)rg * D_ + n0 + col] = (_Float16)acc[r];
            }
        } else {
            // transposed store: Vt[b][d][m]; 8 consecutive m values -> one b128 store
            int b  = rowbase >> 12;
            int m0 = (rowbase & (N_ - 1)) + 8 * hi;
            v8h pk;
            #pragma unroll
            for (int r = 0; r < 8; ++r) pk[r] = (_Float16)acc[r];
            *(v8h*)(Vt + (size_t)b * D_ * N_ + (size_t)(n0 + col) * N_ + m0) = pk;
        }
    }
}

// ---------------------------------------------------------------------------
// Flash attention: out = softmax(K Q^T / sqrt(128)) V   (rows indexed by K)
// grid = (N/16, B); block = 128 (4 waves). All 4 waves share the same 16
// output rows; each covers a disjoint 1024-wide slice of the softmax (m)
// axis, then partials (m,l,O) are combined through LDS.
// ---------------------------------------------------------------------------
__global__ __launch_bounds__(128)
void flash_attn_kernel(const _Float16* __restrict__ Qh,
                       const _Float16* __restrict__ Kh,
                       const _Float16* __restrict__ Vt,
                       float* __restrict__ out)
{
    __shared__ _Float16 plds[4 * 16 * PS];   // per-wave P transpose tiles
    __shared__ float ol[4][16][128];         // per-wave partial O
    __shared__ float ml[4][16], ll[4][16];   // per-wave softmax stats

    const int tid  = threadIdx.x;
    const int wid  = tid >> 5;
    const int lane = tid & 31;
    const int col  = lane & 15;
    const int hi   = lane >> 4;
    const int bb   = blockIdx.y;
    const int n0   = blockIdx.x * 16;        // this block's 16 output rows

    const _Float16* Qb = Qh + (size_t)bb * N_ * D_;
    const _Float16* Kb = Kh + (size_t)bb * N_ * D_;
    const _Float16* Vb = Vt + (size_t)bb * D_ * N_;
    _Float16* pw = plds + wid * 16 * PS;

    // K tile for the block's 16 rows, as 4 A-fragments (resident for the loop)
    F16x16 kfr[4];
    const _Float16* krow = Kb + (size_t)(n0 + col) * D_;
    #pragma unroll
    for (int kc = 0; kc < 4; ++kc) {
        const _Float16* ap = krow + kc * 32 + hi * 8;
        kfr[kc].h[0] = *(const v8h*)ap;
        kfr[kc].h[1] = *(const v8h*)(ap + 16);
    }

    v8f o[8] = {};
    float mrow[8], lrow[8];
    #pragma unroll
    for (int r = 0; r < 8; ++r) { mrow[r] = -1e30f; lrow[r] = 0.0f; }

    const float scale = 0.08838834764831845f;  // 1/sqrt(128)
    const int mbeg = wid * (N_ / 4);
    const int mend = mbeg + (N_ / 4);

    for (int mc = mbeg; mc < mend; mc += 32) {
        // Prefetch next chunk's Q rows and Vt rows into cache while this
        // chunk computes (lowers to global_prefetch_b8; speculative is safe).
        if (mc + 32 < mend) {
            __builtin_prefetch(Qb + (size_t)(mc + 32 + col) * D_, 0, 3);
            __builtin_prefetch(Qb + (size_t)(mc + 48 + col) * D_, 0, 3);
            __builtin_prefetch(Vb + (size_t)(col) * N_ + mc + 32, 0, 3);
            __builtin_prefetch(Vb + (size_t)(32 + col) * N_ + mc + 32, 0, 3);
            __builtin_prefetch(Vb + (size_t)(64 + col) * N_ + mc + 32, 0, 3);
            __builtin_prefetch(Vb + (size_t)(96 + col) * N_ + mc + 32, 0, 3);
        }

        // S = K_tile @ Q^T  for two 16-wide m tiles (k-dim 128 in 4 chunks)
        v8f s0 = {}, s1 = {};
        #pragma unroll
        for (int kc = 0; kc < 4; ++kc) {
            F16x16 q0, q1;
            const _Float16* qp0 = Qb + (size_t)(mc + col) * D_ + kc * 32 + hi * 8;
            const _Float16* qp1 = Qb + (size_t)(mc + 16 + col) * D_ + kc * 32 + hi * 8;
            q0.h[0] = *(const v8h*)qp0; q0.h[1] = *(const v8h*)(qp0 + 16);
            q1.h[0] = *(const v8h*)qp1; q1.h[1] = *(const v8h*)(qp1 + 16);
            s0 = __builtin_amdgcn_wmma_f32_16x16x32_f16(false, kfr[kc].v, false, q0.v,
                                                        (short)0, s0, false, false);
            s1 = __builtin_amdgcn_wmma_f32_16x16x32_f16(false, kfr[kc].v, false, q1.v,
                                                        (short)0, s1, false, false);
        }

        // Online softmax. Row M = r + 8*hi lives across 16 lanes of one half,
        // so reduce with xor masks 1,2,4,8 (stay inside the 16-lane group).
        float alpha[8];
        #pragma unroll
        for (int r = 0; r < 8; ++r) {
            float a0 = s0[r] * scale, a1 = s1[r] * scale;
            float t = fmaxf(a0, a1);
            t = fmaxf(t, __shfl_xor(t, 1, 32));
            t = fmaxf(t, __shfl_xor(t, 2, 32));
            t = fmaxf(t, __shfl_xor(t, 4, 32));
            t = fmaxf(t, __shfl_xor(t, 8, 32));
            float mnew = fmaxf(mrow[r], t);
            float al = __expf(mrow[r] - mnew);
            float p0 = __expf(a0 - mnew);
            float p1 = __expf(a1 - mnew);
            float sm = p0 + p1;
            sm += __shfl_xor(sm, 1, 32);
            sm += __shfl_xor(sm, 2, 32);
            sm += __shfl_xor(sm, 4, 32);
            sm += __shfl_xor(sm, 8, 32);
            lrow[r] = lrow[r] * al + sm;
            mrow[r] = mnew;
            alpha[r] = al;
            s0[r] = p0; s1[r] = p1;
        }
        #pragma unroll
        for (int dt = 0; dt < 8; ++dt)
            #pragma unroll
            for (int r = 0; r < 8; ++r) o[dt][r] *= alpha[r];

        // Transpose P (C layout -> A layout) through this wave's LDS tile.
        // Same-wave LDS ops are kept in order (DScnt), no barrier needed.
        #pragma unroll
        for (int r = 0; r < 8; ++r) {
            pw[(r + 8 * hi) * PS + col]      = (_Float16)s0[r];
            pw[(r + 8 * hi) * PS + 16 + col] = (_Float16)s1[r];
        }
        F16x16 pfr;
        const _Float16* pp = pw + col * PS + hi * 8;
        pfr.h[0] = *(const v8h*)pp;
        pfr.h[1] = *(const v8h*)(pp + 16);

        // O += P @ V : B fragments are contiguous rows of Vt[d][m]
        #pragma unroll
        for (int dt = 0; dt < 8; ++dt) {
            F16x16 vfr;
            const _Float16* vp = Vb + (size_t)(dt * 16 + col) * N_ + mc + hi * 8;
            vfr.h[0] = *(const v8h*)vp;
            vfr.h[1] = *(const v8h*)(vp + 16);
            o[dt] = __builtin_amdgcn_wmma_f32_16x16x32_f16(false, pfr.v, false, vfr.v,
                                                           (short)0, o[dt], false, false);
        }
    }

    // Publish partials, combine across the 4 m-slices, normalize, store.
    #pragma unroll
    for (int dt = 0; dt < 8; ++dt)
        #pragma unroll
        for (int r = 0; r < 8; ++r)
            ol[wid][r + 8 * hi][dt * 16 + col] = o[dt][r];
    if (col == 0) {
        #pragma unroll
        for (int r = 0; r < 8; ++r) {
            ml[wid][r + 8 * hi] = mrow[r];
            ll[wid][r + 8 * hi] = lrow[r];
        }
    }
    __syncthreads();

    float* ob = out + (size_t)bb * N_ * D_;
    #pragma unroll
    for (int R = 0; R < 16; ++R) {
        int dc = tid;  // 128 threads -> 128 d columns, coalesced
        float M = fmaxf(fmaxf(ml[0][R], ml[1][R]), fmaxf(ml[2][R], ml[3][R]));
        float e0 = __expf(ml[0][R] - M);
        float e1 = __expf(ml[1][R] - M);
        float e2 = __expf(ml[2][R] - M);
        float e3 = __expf(ml[3][R] - M);
        float L  = ll[0][R] * e0 + ll[1][R] * e1 + ll[2][R] * e2 + ll[3][R] * e3;
        float v  = ol[0][R][dc] * e0 + ol[1][R][dc] * e1
                 + ol[2][R][dc] * e2 + ol[3][R][dc] * e3;
        ob[(size_t)(n0 + R) * D_ + dc] = v / L;
    }
}

// ---------------------------------------------------------------------------
extern "C" void kernel_launch(void* const* d_in, const int* in_sizes, int n_in,
                              void* d_out, int out_size, void* d_ws, size_t ws_size,
                              hipStream_t stream) {
    (void)in_sizes; (void)n_in; (void)out_size; (void)ws_size;
    const float* h_a = (const float*)d_in[0];
    const float* Wq  = (const float*)d_in[1];
    const float* bq  = (const float*)d_in[2];
    const float* Wk  = (const float*)d_in[3];
    const float* bk  = (const float*)d_in[4];
    const float* Wv  = (const float*)d_in[5];
    const float* bv  = (const float*)d_in[6];
    float* out = (float*)d_out;

    _Float16* Qh = (_Float16*)d_ws;                 // [B*N][128] f16  (4 MB)
    _Float16* Kh = Qh + (size_t)NROW * D_;          // [B*N][128] f16  (4 MB)
    _Float16* Vt = Kh + (size_t)NROW * D_;          // [B][128][N] f16 (4 MB)

    dim3 pgrid(NROW / (16 * ROWT), 3);
    proj_qkv_kernel<<<pgrid, 256, 0, stream>>>(h_a, Wq, bq, Wk, bk, Wv, bv, Qh, Kh, Vt);

    dim3 fgrid(N_ / 16, B_);
    flash_attn_kernel<<<fgrid, 128, 0, stream>>>(Qh, Kh, Vt, out);
}
